// TransparentRNN_11235634447037
// MI455X (gfx1250) — matrix operand order repaired and checked by reading, not compiled, not run on hardware
//
#include <hip/hip_runtime.h>

// Problem sizes (compile-time)
#define BB   32
#define SS   1024
#define DIN  256
#define HH   512
#define H2   1024
#define LL   2
#define DOUT 64
#define LN_EPS 1e-5f

typedef unsigned short u16;
typedef unsigned int   u32;
typedef __attribute__((ext_vector_type(16))) __bf16 v16bf;
typedef __attribute__((ext_vector_type(8)))  float  v8f;

// Detect TDM builtins (device pass only; host pass takes fallback).
#if defined(__has_builtin)
#if __has_builtin(__builtin_amdgcn_tensor_load_to_lds) && \
    __has_builtin(__builtin_amdgcn_s_wait_tensorcnt)
#define USE_TDM 1
#endif
#endif
#ifndef USE_TDM
#define USE_TDM 0
#endif

struct Frag { union { v16bf v; uint4 q[2]; }; };

// Hardware bf16 converts (v_cvt_pk_bf16_f32) instead of manual rounding.
__device__ inline u32 pack2(float a, float b) {
  union { struct { __bf16 lo, hi; } h; u32 u; } t;
  t.h.lo = (__bf16)a;
  t.h.hi = (__bf16)b;
  return t.u;
}
__device__ inline u16 f32_to_bf16(float f) {
  union { __bf16 h; u16 u; } t;
  t.h = (__bf16)f;
  return t.u;
}
__device__ inline float sigf(float x) { return 1.f / (1.f + __expf(-x)); }

// ---- WMMA fragment loaders (CDNA5 16-bit layouts, wave32) ----
// A (16x32 bf16) from split LDS staging: abx = K[0,512), abh = K[512,1024),
// each 32 rows x 512 cols row-major (32 KB, contiguous -> TDM friendly).
// Fragments never straddle the 512 boundary (k-tile aligned).
__device__ inline v16bf load_a_lds(const u16* abx, const u16* abh, int mtile,
                                   int kt, int lane) {
  int m   = mtile * 16 + (lane & 15);
  int k0g = kt * 32 + (lane >> 4) * 8;      // lanes 16-31 start at K+8
  const u16* base = (k0g < HH) ? abx : abh;
  int k0 = k0g & (HH - 1);
  const u16* p = base + m * HH + k0;
  Frag f;
  f.q[0] = *(const uint4*)(p);              // K = k0 .. k0+7
  f.q[1] = *(const uint4*)(p + 16);         // K = k0+16 .. k0+23
  return f.v;
}
// B (32x16 bf16) from global, weights pre-packed N-major (N x K):
// each lane reads K = k0..k0+7 and k0+16..k0+23 (2 x 16 bytes).
__device__ inline v16bf load_b_gl(const u16* wt, int K, int ntile, int kt,
                                  int lane) {
  int n  = ntile * 16 + (lane & 15);
  int k0 = kt * 32 + (lane >> 4) * 8;       // lanes 16-31 start at K+8
  const u16* p = wt + (size_t)n * K + k0;
  Frag f;
  f.q[0] = *(const uint4*)(p);
  f.q[1] = *(const uint4*)(p + 16);
  return f.v;
}
__device__ inline v8f wmma_bf16(v16bf a, v16bf b, v8f c) {
  return __builtin_amdgcn_wmma_f32_16x16x32_bf16(false, a, false, b, (short)0,
                                                 c, false, false);
}

// ------------------------------------------------------------------
// Kernel 1: pack/convert all weights to bf16, transposed to N-major.
// ------------------------------------------------------------------
#define RI  (HH * DIN)
#define RZR (LL * H2 * H2)
#define RC  (LL * HH * H2)

__global__ void pack_weights(const float* __restrict__ W_in,
                             const float* __restrict__ Wz,
                             const float* __restrict__ Wr,
                             const float* __restrict__ Wc,
                             u16* __restrict__ wtin, u16* __restrict__ wzr,
                             u16* __restrict__ wc) {
  int i = blockIdx.x * blockDim.x + threadIdx.x;
  if (i < RI) {
    int n = i / DIN, k = i % DIN;
    wtin[i] = f32_to_bf16(W_in[k * HH + n]);
  } else if (i < RI + RZR) {
    int j = i - RI;
    int l = j / (H2 * H2);
    int rem = j % (H2 * H2);
    int n = rem / H2, k = rem % H2;
    float v = (n < HH) ? Wz[(size_t)l * H2 * HH + (size_t)k * HH + n]
                       : Wr[(size_t)l * H2 * HH + (size_t)k * HH + (n - HH)];
    wzr[j] = f32_to_bf16(v);
  } else if (i < RI + RZR + RC) {
    int j = i - RI - RZR;
    int l = j / (HH * H2);
    int rem = j % (HH * H2);
    int n = rem / H2, k = rem % H2;
    wc[j] = f32_to_bf16(Wc[(size_t)l * H2 * HH + (size_t)k * HH + n]);
  }
}

// ------------------------------------------------------------------
// Kernel 2: input projection  x_tr = x @ W_in + b_in  (bf16 WMMA, f32 acc)
// output stored (S,B,HH) bf16 so each timestep is one contiguous 32 KB tile.
// ------------------------------------------------------------------
__global__ __launch_bounds__(256) void input_proj(const float* __restrict__ x,
                                                  const u16* __restrict__ wtin,
                                                  const float* __restrict__ b_in,
                                                  u16* __restrict__ xtr) {
  int gwave = blockIdx.x * (blockDim.x >> 5) + (threadIdx.x >> 5);
  int lane  = threadIdx.x & 31;
  int mt = gwave >> 5;        // 0..2047  (M = 32768 rows)
  int nt = gwave & 31;        // 0..31    (N = 512 cols)

  v8f acc = {};
  int m  = mt * 16 + (lane & 15);
  int hs = lane >> 4;
  for (int kt = 0; kt < DIN / 32; ++kt) {
    int k0 = kt * 32 + hs * 8;
    const float* p = x + (size_t)m * DIN + k0;
    float4 f0 = *(const float4*)(p);
    float4 f1 = *(const float4*)(p + 4);
    float4 f2 = *(const float4*)(p + 16);
    float4 f3 = *(const float4*)(p + 20);
    Frag a;
    a.q[0] = make_uint4(pack2(f0.x, f0.y), pack2(f0.z, f0.w),
                        pack2(f1.x, f1.y), pack2(f1.z, f1.w));
    a.q[1] = make_uint4(pack2(f2.x, f2.y), pack2(f2.z, f2.w),
                        pack2(f3.x, f3.y), pack2(f3.z, f3.w));
    v16bf b = load_b_gl(wtin, DIN, nt, kt, lane);
    acc = wmma_bf16(a.v, b, acc);
  }
  int col  = nt * 16 + (lane & 15);
  float bias = b_in[col];
  int mb = hs * 8;
  for (int r = 0; r < 8; ++r) {
    int row = mt * 16 + r + mb;             // flattened (b*S + s)
    int bidx = row >> 10, sidx = row & (SS - 1);
    xtr[((size_t)sidx * BB + bidx) * HH + col] = f32_to_bf16(acc[r] + bias);
  }
}

// ------------------------------------------------------------------
// Kernel 3: persistent single-workgroup GRU scan.
// LDS (256 KB of the 320 KB WGP LDS):
//   abx (32x512 bf16, x_t / first K half)   32 KB  <- filled by TDM
//   abh (32x512 bf16, h / r*h second half)  32 KB
//   G   (32x1024 f32: z | r -> h~)         128 KB
//   hbuf(32x512 f32 hidden)                 64 KB
// ------------------------------------------------------------------
__global__ __launch_bounds__(1024, 1) void rnn_scan(
    const u16*  __restrict__ xtr,     // (S,B,H) bf16
    const u16*  __restrict__ wzr,     // (L, H2 x H2) bf16 N-major
    const u16*  __restrict__ wc,      // (L, HH x H2) bf16 N-major
    const float* __restrict__ hidden0,
    const float* __restrict__ bz, const float* __restrict__ br,
    const float* __restrict__ bc,
    const float* __restrict__ lng, const float* __restrict__ lnb,
    float* __restrict__ out_hidden,   // (B,S,H)
    float* __restrict__ out_final) {  // (B,H)
  extern __shared__ char smem[];
  u16*   abx  = (u16*)smem;                          //  32 KB @ 0
  u16*   abh  = (u16*)(smem + 32 * 1024);            //  32 KB
  float* G    = (float*)(smem + 64 * 1024);          // 128 KB
  float* hbuf = (float*)(smem + 192 * 1024);         //  64 KB

  const int tid  = threadIdx.x;
  const int lane = tid & 31;
  const int wave = tid >> 5;

  for (int i = tid; i < BB * HH; i += 1024) hbuf[i] = hidden0[i];
  __syncthreads();

  for (int t = 0; t < SS; ++t) {
    const u16* xt = xtr + (size_t)t * BB * HH;

    // ---- Stage A operand: abx = x_t (TDM 1D tile), abh = bf16(h) ----
#if USE_TDM
    if (wave == 0) {
      typedef __attribute__((ext_vector_type(4))) unsigned int tdm_u4;
      typedef __attribute__((ext_vector_type(8))) int tdm_i8;
      typedef __attribute__((ext_vector_type(4))) int tdm_i4;
      unsigned long long ga = (unsigned long long)(uintptr_t)xt;
      // D# group 0: count=1 | lds_addr=0 (abx) | global_addr | type=2
      tdm_u4 g0 = {1u, 0u, (unsigned)(ga & 0xffffffffu),
                   (unsigned)((ga >> 32) & 0x1ffffffu) | (2u << 30)};
      // D# group 1: data_size=2B; 1-D copy of 16384 elements (32 KB).
      // tensor_dim0=16384 (bits79:48), tensor_dim1=1 (bits111:80),
      // tile_dim0=16384 (bits127:112), tile_dim1=1 (bits143:128),
      // tensor_dim0_stride=16384 (bits207:160).
      tdm_i8 g1 = {(int)(1u << 16),                 // data_size code 1 (2B)
                   (int)(16384u << 16),             // tensor_dim0 lo16
                   (int)(1u << 16),                 // dim0 hi16=0 | dim1 lo16=1
                   (int)(16384u << 16),             // dim1 hi16=0 | tile_dim0
                   (int)1,                          // tile_dim1=1, tile_dim2=0
                   (int)16384,                      // dim0_stride lo32
                   0, 0};
      tdm_i4 g2 = {0, 0, 0, 0};
      tdm_i4 g3 = {0, 0, 0, 0};
      tdm_i8 g4 = {0, 0, 0, 0, 0, 0, 0, 0};       // 6-arg form (clang-23 lane)
      __builtin_amdgcn_tensor_load_to_lds(g0, g1, g2, g3, g4, 0);
    }
#else
    {
      u32* dx = (u32*)abx;
      const u32* sx = (const u32*)xt;
      for (int i = tid; i < BB * HH / 2; i += 1024) dx[i] = sx[i];
    }
#endif
    {
      u32* dh = (u32*)abh;
      for (int i = tid; i < BB * HH / 2; i += 1024) {
        float a = hbuf[2 * i], b = hbuf[2 * i + 1];
        dh[i] = pack2(a, b);
      }
    }
#if USE_TDM
    if (wave == 0) __builtin_amdgcn_s_wait_tensorcnt(0);
#endif
    __syncthreads();

    for (int l = 0; l < LL; ++l) {
      const u16* Wzrl = wzr + (size_t)l * H2 * H2;
      const u16* Wcl  = wc  + (size_t)l * HH * H2;

      // ---- GEMM1: gates z|r = sigmoid([x_t,h] @ [Wz|Wr] + [bz|br]) ----
      {
        v8f a00 = {}, a01 = {}, a10 = {}, a11 = {};
        const int nt0 = wave * 2, nt1 = wave * 2 + 1;
        for (int kt = 0; kt < 32; ++kt) {
          v16bf af0 = load_a_lds(abx, abh, 0, kt, lane);
          v16bf af1 = load_a_lds(abx, abh, 1, kt, lane);
          v16bf bf0 = load_b_gl(Wzrl, H2, nt0, kt, lane);
          v16bf bf1 = load_b_gl(Wzrl, H2, nt1, kt, lane);
          if (kt + 4 < 32) {  // stream weights from L2
            __builtin_prefetch(Wzrl + (size_t)(nt0 * 16 + (lane & 15)) * H2 +
                               (kt + 4) * 32, 0, 1);
          }
          a00 = wmma_bf16(af0, bf0, a00);
          a10 = wmma_bf16(af1, bf0, a10);
          a01 = wmma_bf16(af0, bf1, a01);
          a11 = wmma_bf16(af1, bf1, a11);
        }
        int c0 = nt0 * 16 + (lane & 15), c1 = nt1 * 16 + (lane & 15);
        float bia0 = (c0 < HH) ? bz[l * HH + c0] : br[l * HH + c0 - HH];
        float bia1 = (c1 < HH) ? bz[l * HH + c1] : br[l * HH + c1 - HH];
        int mb = (lane >> 4) * 8;
        for (int r = 0; r < 8; ++r) {
          int m0 = r + mb, m1 = 16 + r + mb;
          G[m0 * H2 + c0] = sigf(a00[r] + bia0);
          G[m1 * H2 + c0] = sigf(a10[r] + bia0);
          G[m0 * H2 + c1] = sigf(a01[r] + bia1);
          G[m1 * H2 + c1] = sigf(a11[r] + bia1);
        }
      }
      __syncthreads();

      // ---- candidate A operand: abh = bf16(r * h) ----
      {
        u32* dh = (u32*)abh;
        for (int i = tid; i < BB * HH / 2; i += 1024) {
          int b = i >> 8, jj = i & 255;       // 256 u32 per row
          int j = jj * 2;
          float r0 = G[b * H2 + HH + j]     * hbuf[b * HH + j];
          float r1 = G[b * H2 + HH + j + 1] * hbuf[b * HH + j + 1];
          dh[i] = pack2(r0, r1);
        }
      }
      __syncthreads();

      // ---- GEMM2: h~ = tanh([x_t, r*h] @ Wc + bc), stored in r's slot ----
      {
        v8f a0 = {}, a1 = {};
        const int nt = wave;  // 32 ntiles cover N = 512
        for (int kt = 0; kt < 32; ++kt) {
          v16bf af0 = load_a_lds(abx, abh, 0, kt, lane);
          v16bf af1 = load_a_lds(abx, abh, 1, kt, lane);
          v16bf bf0 = load_b_gl(Wcl, H2, nt, kt, lane);
          a0 = wmma_bf16(af0, bf0, a0);
          a1 = wmma_bf16(af1, bf0, a1);
        }
        int c = nt * 16 + (lane & 15);
        float bia = bc[l * HH + c];
        int mb = (lane >> 4) * 8;
        for (int r = 0; r < 8; ++r) {
          G[(r + mb) * H2 + HH + c]      = tanhf(a0[r] + bia);
          G[(16 + r + mb) * H2 + HH + c] = tanhf(a1[r] + bia);
        }
      }
      __syncthreads();

      // ---- blend + LayerNorm: one wave per batch row ----
      {
        const int b = wave;
        float s1 = 0.f, s2 = 0.f;
        for (int j = lane; j < HH; j += 32) {
          float z  = G[b * H2 + j];
          float ht = G[b * H2 + HH + j];
          float h  = hbuf[b * HH + j];
          float hn = (1.f - z) * h + z * ht;
          hbuf[b * HH + j] = hn;
          s1 += hn; s2 += hn * hn;
        }
        for (int off = 16; off; off >>= 1) {
          s1 += __shfl_xor(s1, off, 32);
          s2 += __shfl_xor(s2, off, 32);
        }
        float mean = s1 * (1.f / HH);
        float var  = s2 * (1.f / HH) - mean * mean;
        float inv  = rsqrtf(var + LN_EPS);
        for (int j = lane; j < HH; j += 32) {
          float v = (hbuf[b * HH + j] - mean) * inv * lng[l * HH + j] + lnb[l * HH + j];
          hbuf[b * HH + j] = v;
          if (l == 0) {  // layer 1 sees combined = [h0, h0]
            u16 bv = f32_to_bf16(v);
            abx[b * HH + j] = bv;
            abh[b * HH + j] = bv;
          } else {
            out_hidden[(size_t)b * SS * HH + (size_t)t * HH + j] = v;
          }
        }
      }
      __syncthreads();
    }
  }
  for (int i = tid; i < BB * HH; i += 1024) out_final[i] = hbuf[i];
}

// ------------------------------------------------------------------
// Kernel 4: output heads (tiny GEMV + sigmoid) from final hidden.
// ------------------------------------------------------------------
__global__ __launch_bounds__(128) void heads(const float* __restrict__ last,
                                             const float* __restrict__ Wb,
                                             const float* __restrict__ bb,
                                             const float* __restrict__ Wa,
                                             const float* __restrict__ ba,
                                             float* __restrict__ beliefs,
                                             float* __restrict__ actions) {
  int b = blockIdx.x, tid = threadIdx.x;
  if (tid < DOUT) {
    float s = 0.f;
    for (int j = 0; j < HH; ++j) s += last[b * HH + j] * Wb[j * DOUT + tid];
    beliefs[b * DOUT + tid] = sigf(s + bb[tid]);
  } else if (tid == DOUT) {
    float s = 0.f;
    for (int j = 0; j < HH; ++j) s += last[b * HH + j] * Wa[j];
    actions[b] = sigf(s + ba[0]);
  }
}

// ------------------------------------------------------------------
extern "C" void kernel_launch(void* const* d_in, const int* in_sizes, int n_in,
                              void* d_out, int out_size, void* d_ws, size_t ws_size,
                              hipStream_t stream) {
  const float* x      = (const float*)d_in[0];
  const float* hidden = (const float*)d_in[1];
  const float* W_in   = (const float*)d_in[2];
  const float* b_in   = (const float*)d_in[3];
  const float* Wz     = (const float*)d_in[4];
  const float* bz     = (const float*)d_in[5];
  const float* Wr     = (const float*)d_in[6];
  const float* br     = (const float*)d_in[7];
  const float* Wc     = (const float*)d_in[8];
  const float* bc     = (const float*)d_in[9];
  const float* lng    = (const float*)d_in[10];
  const float* lnb    = (const float*)d_in[11];
  const float* W_bel  = (const float*)d_in[12];
  const float* b_bel  = (const float*)d_in[13];
  const float* W_act  = (const float*)d_in[14];
  const float* b_act  = (const float*)d_in[15];

  char* ws = (char*)d_ws;
  u16* wtin = (u16*)ws;                                     // 256 KB
  u16* wzr  = (u16*)(ws + 262144);                          //   4 MB
  u16* wcd  = (u16*)(ws + 262144 + 4194304);                //   2 MB
  u16* xtr  = (u16*)(ws + 262144 + 4194304 + 2097152);      //  32 MB

  float* out         = (float*)d_out;
  float* out_hidden  = out;                                 // B*S*H
  float* out_beliefs = out + (size_t)BB * SS * HH;          // B*DOUT
  float* out_actions = out_beliefs + BB * DOUT;             // B
  float* out_final   = out_actions + BB;                    // B*H

  int total_pack = RI + RZR + RC;
  pack_weights<<<(total_pack + 255) / 256, 256, 0, stream>>>(
      W_in, Wz, Wr, Wc, wtin, wzr, wcd);

  // 2048 m-tiles * 32 n-tiles = 65536 waves / 8 waves-per-block
  input_proj<<<8192, 256, 0, stream>>>(x, wtin, b_in, xtr);

  rnn_scan<<<1, 1024, 262144, stream>>>(xtr, wzr, wcd, hidden,
                                        bz, br, bc, lng, lnb,
                                        out_hidden, out_final);

  heads<<<BB, 128, 0, stream>>>(out_final, W_bel, b_bel, W_act, b_act,
                                out_beliefs, out_actions);
}
